// SMPLHPyTorchWrapper_28845000360698
// MI455X (gfx1250) — matrix operand order repaired and checked by reading, not compiled, not run on hardware
//
#include <hip/hip_runtime.h>
#include <math.h>

// ---------------- problem constants ----------------
#define BATCH   512
#define NJ      52
#define NV      6890
#define NB      10
#define NP      459            // (NJ-1)*9
#define VK      20670          // NV*3
#define KPOSE   480            // 459 pose + 10 betas + 11 pad  (15 * 32)
#define KSTEPS  15
#define NTILES  1292           // ceil(20670/16)
#define WTILES  431            // ceil(6890/16)
#define MW      6896           // WTILES*16

// ---------------- output layout (floats) ----------------
#define OUT_VERTS  ((size_t)0)
#define OUT_JTR    ((size_t)BATCH * VK)                    // 10,583,040
#define OUT_VPOSED (OUT_JTR + (size_t)BATCH * NJ * 3)      // 10,662,912
#define OUT_NAKED  (OUT_VPOSED + (size_t)BATCH * VK)       // 21,245,952

// ---------------- workspace layout (bytes) ----------------
#define O_JT    ((size_t)0)          // 156 f32
#define O_JS    ((size_t)1024)       // 1560 f32
#define O_ROT   ((size_t)8192)       // 512*52*9 f32
#define O_THJ   ((size_t)983040)     // 512*156 f32
#define O_A     ((size_t)1310720)    // 512*52*12 f32
#define O_A2    ((size_t)2621440)    // 512*52*16 f32
#define O_AH    ((size_t)4325376)    // 512*480 f16   (pose GEMM A, row-major)
#define O_WH    ((size_t)4816896)    // 6896*64 f16   (skin GEMM A, row-major padded)
#define O_A2H   ((size_t)5699584)    // 512*2*32*16 f16 (skin GEMM B, fragment-swizzled)
#define O_DH    ((size_t)6748160)    // 1292*15*32*16 f16 (pose GEMM B, fragment-swizzled)

typedef __attribute__((ext_vector_type(16))) _Float16 v16h;
typedef __attribute__((ext_vector_type(8)))  _Float16 v8h;
typedef __attribute__((ext_vector_type(8)))  float    v8f;

union FragH { v16h v; v8h h[2]; };

__constant__ int d_parents[NJ] = {
  -1, 0, 0, 0, 1, 2, 3, 4, 5, 6, 7, 8, 9, 9, 9, 12, 13, 14, 16, 17, 18, 19,
  20, 22, 23, 20, 25, 26, 20, 28, 29, 20, 31, 32, 20, 34, 35, 21, 37, 38, 21,
  40, 41, 21, 43, 44, 21, 46, 47, 21, 49, 50
};

// ---------------------------------------------------------------------------
// JT[j,k] = sum_v Jreg[j,v]*v_template[v,k];  JS[j,k,nb] = sum_v Jreg[j,v]*shapedirs[v,k,nb]
__global__ __launch_bounds__(256) void k_jointreg(
    const float* __restrict__ Jreg, const float* __restrict__ vtempl,
    const float* __restrict__ shaped, float* __restrict__ JT, float* __restrict__ JS)
{
  int c = blockIdx.x * blockDim.x + threadIdx.x;
  if (c >= 156 + 1560) return;
  if (c < 156) {
    int j = c / 3, k = c % 3;
    const float* Jr = Jreg + (size_t)j * NV;
    float s = 0.f;
    for (int v = 0; v < NV; ++v) s += Jr[v] * vtempl[v * 3 + k];
    JT[c] = s;
  } else {
    int c2 = c - 156;
    int j = c2 / 30, rem = c2 % 30, k = rem / 10, nb = rem % 10;
    const float* Jr = Jreg + (size_t)j * NV;
    float s = 0.f;
    for (int v = 0; v < NV; ++v) s += Jr[v] * shaped[(size_t)(v * 3 + k) * NB + nb];
    JS[c2] = s;
  }
}

// ---------------------------------------------------------------------------
// Rodrigues for every (b,j); writes rot (f32 ws) and the pose-GEMM A row (f16).
__global__ __launch_bounds__(256) void k_rodrigues(
    const float* __restrict__ pose, const float* __restrict__ betas,
    float* __restrict__ rot, _Float16* __restrict__ Ah)
{
  int t = blockIdx.x * blockDim.x + threadIdx.x;
  if (t >= BATCH * NJ) return;
  int b = t / NJ, j = t % NJ;
  float rx = pose[b * NJ * 3 + j * 3 + 0];
  float ry = pose[b * NJ * 3 + j * 3 + 1];
  float rz = pose[b * NJ * 3 + j * 3 + 2];
  float th = sqrtf(rx * rx + ry * ry + rz * rz + 1e-8f);
  float inv = 1.0f / th;
  float x = rx * inv, y = ry * inv, z = rz * inv;
  float s = sinf(th), c = cosf(th), C = 1.0f - c;
  float R[9];
  R[0] = 1.0f - C * (y * y + z * z);
  R[1] = -s * z + C * (x * y);
  R[2] =  s * y + C * (x * z);
  R[3] =  s * z + C * (x * y);
  R[4] = 1.0f - C * (x * x + z * z);
  R[5] = -s * x + C * (y * z);
  R[6] = -s * y + C * (x * z);
  R[7] =  s * x + C * (y * z);
  R[8] = 1.0f - C * (x * x + y * y);
  float* rd = rot + (size_t)b * NJ * 9 + j * 9;
  #pragma unroll
  for (int e = 0; e < 9; ++e) rd[e] = R[e];
  _Float16* ar = Ah + (size_t)b * KPOSE;
  if (j > 0) {
    #pragma unroll
    for (int e = 0; e < 9; ++e)
      ar[(j - 1) * 9 + e] = (_Float16)(R[e] - ((e % 4 == 0) ? 1.0f : 0.0f));
  } else {
    for (int nb = 0; nb < NB; ++nb) ar[NP + nb] = (_Float16)betas[b * NB + nb];
    for (int q = NP + NB; q < KPOSE; ++q) ar[q] = (_Float16)0.0f;
  }
}

// ---------------------------------------------------------------------------
// Pose-GEMM B operand, pre-swizzled into the B-fragment k order used here (contiguous per lane half):
// Dh[((ntile*15+ks)*32 + lane)*16 + i]  <-  D[k = ks*32 + (lane<16?0:16) + i][vk = ntile*16 + lane%16]
// where D rows 0..458 = posedirs[vk][k], 459..468 = shapedirs[vk][k-459], rest 0.
__global__ __launch_bounds__(256) void k_build_dh(
    const float* __restrict__ posedirs, const float* __restrict__ shaped,
    _Float16* __restrict__ Dh)
{
  int tid = blockIdx.x * blockDim.x + threadIdx.x;
  if (tid >= NTILES * KSTEPS * 32) return;
  int lane  = tid & 31;
  int ks    = (tid >> 5) % KSTEPS;
  int ntile = tid / (KSTEPS * 32);
  int vk = ntile * 16 + (lane & 15);
  int kb = ks * 32 + ((lane & 16) ? 16 : 0);
  _Float16* dst = Dh + (size_t)tid * 16;
  #pragma unroll
  for (int i = 0; i < 16; ++i) {
    int k = kb + i;
    float val = 0.f;
    if (vk < VK) {
      if (k < NP)           val = posedirs[(size_t)vk * NP + k];
      else if (k < NP + NB) val = shaped[(size_t)vk * NB + (k - NP)];
    }
    dst[i] = (_Float16)val;
  }
}

// ---------------------------------------------------------------------------
// Skin-GEMM A operand: weights padded to [6896][64], row-major f16.
__global__ __launch_bounds__(256) void k_build_wh(
    const float* __restrict__ weights, _Float16* __restrict__ Wh)
{
  int t = blockIdx.x * blockDim.x + threadIdx.x;
  if (t >= MW * 64) return;
  int v = t >> 6, j = t & 63;
  float val = (v < NV && j < NJ) ? weights[(size_t)v * NJ + j] : 0.f;
  Wh[t] = (_Float16)val;
}

// ---------------------------------------------------------------------------
// Kinematic chain, one thread per batch. Produces A (f32 ws), A2 (f32 ws), th_jtr out.
__global__ __launch_bounds__(256) void k_chain(
    const float* __restrict__ JT, const float* __restrict__ JS,
    const float* __restrict__ betas, const float* __restrict__ trans,
    const float* __restrict__ rot, float* __restrict__ thj,
    float* __restrict__ Aw, float* __restrict__ A2w, float* __restrict__ out)
{
  int b = blockIdx.x * blockDim.x + threadIdx.x;
  if (b >= BATCH) return;
  const float* bt = betas + (size_t)b * NB;
  float* tj = thj + (size_t)b * NJ * 3;
  for (int c = 0; c < NJ * 3; ++c) {
    float s = JT[c];
    #pragma unroll
    for (int nb = 0; nb < NB; ++nb) s += JS[c * NB + nb] * bt[nb];
    tj[c] = s;
  }
  float* A = Aw + (size_t)b * NJ * 12;
  const float* R = rot + (size_t)b * NJ * 9;
  #pragma unroll
  for (int m = 0; m < 3; ++m) {
    A[m * 4 + 0] = R[m * 3 + 0];
    A[m * 4 + 1] = R[m * 3 + 1];
    A[m * 4 + 2] = R[m * 3 + 2];
    A[m * 4 + 3] = tj[m];
  }
  for (int i = 1; i < NJ; ++i) {
    int p = d_parents[i];
    const float* Pa = A + p * 12;
    const float* Ri = R + i * 9;
    float t0 = tj[i * 3 + 0] - tj[p * 3 + 0];
    float t1 = tj[i * 3 + 1] - tj[p * 3 + 1];
    float t2 = tj[i * 3 + 2] - tj[p * 3 + 2];
    float* Ai = A + i * 12;
    #pragma unroll
    for (int m = 0; m < 3; ++m) {
      float p0 = Pa[m * 4 + 0], p1 = Pa[m * 4 + 1], p2 = Pa[m * 4 + 2], p3 = Pa[m * 4 + 3];
      Ai[m * 4 + 0] = p0 * Ri[0] + p1 * Ri[3] + p2 * Ri[6];
      Ai[m * 4 + 1] = p0 * Ri[1] + p1 * Ri[4] + p2 * Ri[7];
      Ai[m * 4 + 2] = p0 * Ri[2] + p1 * Ri[5] + p2 * Ri[8];
      Ai[m * 4 + 3] = p0 * t0 + p1 * t1 + p2 * t2 + p3;
    }
  }
  float tx = trans[b * 3 + 0], ty = trans[b * 3 + 1], tz = trans[b * 3 + 2];
  for (int j = 0; j < NJ; ++j) {
    out[OUT_JTR + (size_t)b * NJ * 3 + j * 3 + 0] = A[j * 12 + 0 * 4 + 3] + tx;
    out[OUT_JTR + (size_t)b * NJ * 3 + j * 3 + 1] = A[j * 12 + 1 * 4 + 3] + ty;
    out[OUT_JTR + (size_t)b * NJ * 3 + j * 3 + 2] = A[j * 12 + 2 * 4 + 3] + tz;
  }
  float* A2 = A2w + (size_t)b * NJ * 16;
  for (int j = 0; j < NJ; ++j) {
    float jx = tj[j * 3 + 0], jy = tj[j * 3 + 1], jz = tj[j * 3 + 2];
    #pragma unroll
    for (int m = 0; m < 3; ++m) {
      float a0 = A[j * 12 + m * 4 + 0];
      float a1 = A[j * 12 + m * 4 + 1];
      float a2 = A[j * 12 + m * 4 + 2];
      float a3 = A[j * 12 + m * 4 + 3];
      float tmp = a0 * jx + a1 * jy + a2 * jz;
      A2[j * 16 + m * 4 + 0] = a0;
      A2[j * 16 + m * 4 + 1] = a1;
      A2[j * 16 + m * 4 + 2] = a2;
      A2[j * 16 + m * 4 + 3] = a3 - tmp;
    }
    A2[j * 16 + 12] = 0.f; A2[j * 16 + 13] = 0.f; A2[j * 16 + 14] = 0.f; A2[j * 16 + 15] = 1.f;
  }
}

// ---------------------------------------------------------------------------
// Skin-GEMM B operand: A2 swizzled to WMMA f16 B-fragment layout [b][2][32][16].
__global__ __launch_bounds__(256) void k_build_a2h(
    const float* __restrict__ A2w, _Float16* __restrict__ A2h)
{
  int t = blockIdx.x * blockDim.x + threadIdx.x;
  if (t >= BATCH * 64) return;
  int lane = t & 31;
  int ks   = (t >> 5) & 1;
  int b    = t >> 6;
  int n  = lane & 15;
  int kb = ks * 32 + ((lane & 16) ? 16 : 0);
  _Float16* dst = A2h + (size_t)t * 16;
  const float* src = A2w + (size_t)b * NJ * 16;
  #pragma unroll
  for (int i = 0; i < 16; ++i) {
    int j = kb + i;
    dst[i] = (_Float16)(j < NJ ? src[j * 16 + n] : 0.f);
  }
}

// ---------------------------------------------------------------------------
// Big pose GEMM: C[b, vk] = sum_k Ah[b,k]*D[k,vk]; epilogue adds v_template.
// One block per N-tile; 8 waves x 4 batch-tiles cover all 512 batches so the
// 20MB B panel streams from L2/HBM exactly once per N-tile column.
__global__ __launch_bounds__(256) void k_gemm_pose(
    const _Float16* __restrict__ Ah, const _Float16* __restrict__ Dh,
    const float* __restrict__ vtempl, float* __restrict__ out)
{
  int ntile = blockIdx.x;
  int lane = threadIdx.x & 31;
  int wave = threadIdx.x >> 5;
  int col = lane & 15;
  int hi  = lane >> 4;
  v8f acc[4] = {};
  for (int ks = 0; ks < KSTEPS; ++ks) {
    FragH bf;
    const _Float16* bp = Dh + ((size_t)(ntile * KSTEPS + ks) * 32 + lane) * 16;
    bf.h[0] = *(const v8h*)(bp);
    bf.h[1] = *(const v8h*)(bp + 8);
    #pragma unroll
    for (int t = 0; t < 4; ++t) {
      int brow = (wave * 4 + t) * 16 + col;
      const _Float16* ap = Ah + (size_t)brow * KPOSE + ks * 32 + hi * 16;
      FragH af;
      af.h[0] = *(const v8h*)(ap);
      af.h[1] = *(const v8h*)(ap + 8);
      acc[t] = __builtin_amdgcn_wmma_f32_16x16x32_f16(
          false, af.v, false, bf.v, (short)0, acc[t], false, false);
    }
  }
  int vk = ntile * 16 + col;
  if (vk < VK) {
    float vt = vtempl[vk];
    #pragma unroll
    for (int t = 0; t < 4; ++t) {
      int bbase = (wave * 4 + t) * 16 + hi * 8;
      #pragma unroll
      for (int r = 0; r < 8; ++r) {
        float val = acc[t][r] + vt;
        size_t o = (size_t)(bbase + r) * VK + vk;
        out[OUT_VPOSED + o] = val;
        out[OUT_NAKED + o]  = val;
      }
    }
  }
}

// ---------------------------------------------------------------------------
// Skinning: per (b, v-tile) WMMA gives T4[v,(m,n)]; 4-lane shfl reduction vs
// rest_h produces th_verts directly (th_T never materialized).
__global__ __launch_bounds__(256) void k_skin(
    const _Float16* __restrict__ Wh, const _Float16* __restrict__ A2h,
    const float* __restrict__ vposed, const float* __restrict__ trans,
    float* __restrict__ out)
{
  int waveId = blockIdx.x * 8 + (threadIdx.x >> 5);
  int lane = threadIdx.x & 31;
  int b  = waveId / WTILES;
  int vt = waveId % WTILES;
  int col = lane & 15;
  int hi  = lane >> 4;
  v8f acc = {};
  #pragma unroll
  for (int ks = 0; ks < 2; ++ks) {
    FragH af, bf;
    const _Float16* ap = Wh + (size_t)(vt * 16 + col) * 64 + ks * 32 + hi * 16;
    af.h[0] = *(const v8h*)(ap);
    af.h[1] = *(const v8h*)(ap + 8);
    const _Float16* bp = A2h + ((size_t)(b * 2 + ks) * 32 + lane) * 16;
    bf.h[0] = *(const v8h*)(bp);
    bf.h[1] = *(const v8h*)(bp + 8);
    acc = __builtin_amdgcn_wmma_f32_16x16x32_f16(
        false, af.v, false, bf.v, (short)0, acc, false, false);
  }
  int n = col & 3;
  int m = col >> 2;
  float tb = trans[b * 3 + (m < 3 ? m : 0)];
  #pragma unroll
  for (int r = 0; r < 8; ++r) {
    int v = vt * 16 + r + hi * 8;
    int vc = v < NV ? v : NV - 1;
    float coeff = (n == 3) ? 1.0f : vposed[(size_t)b * VK + vc * 3 + n];
    float tpart = acc[r] * coeff;
    tpart += __shfl_xor(tpart, 1, 32);
    tpart += __shfl_xor(tpart, 2, 32);
    if (n == 0 && m < 3 && v < NV)
      out[(size_t)b * VK + v * 3 + m] = tpart + tb;
  }
}

// ---------------------------------------------------------------------------
extern "C" void kernel_launch(void* const* d_in, const int* in_sizes, int n_in,
                              void* d_out, int out_size, void* d_ws, size_t ws_size,
                              hipStream_t stream) {
  const float* pose   = (const float*)d_in[0];
  const float* betas  = (const float*)d_in[1];
  const float* trans  = (const float*)d_in[2];
  const float* vtempl = (const float*)d_in[3];
  const float* shaped = (const float*)d_in[4];
  const float* posed  = (const float*)d_in[5];
  const float* Jreg   = (const float*)d_in[6];
  const float* wgt    = (const float*)d_in[7];
  float* out = (float*)d_out;
  char* ws = (char*)d_ws;

  float*     JT  = (float*)(ws + O_JT);
  float*     JS  = (float*)(ws + O_JS);
  float*     rot = (float*)(ws + O_ROT);
  float*     thj = (float*)(ws + O_THJ);
  float*     Aw  = (float*)(ws + O_A);
  float*     A2w = (float*)(ws + O_A2);
  _Float16*  Ah  = (_Float16*)(ws + O_AH);
  _Float16*  Wh  = (_Float16*)(ws + O_WH);
  _Float16*  A2h = (_Float16*)(ws + O_A2H);
  _Float16*  Dh  = (_Float16*)(ws + O_DH);

  k_jointreg<<<7, 256, 0, stream>>>(Jreg, vtempl, shaped, JT, JS);
  k_rodrigues<<<(BATCH * NJ + 255) / 256, 256, 0, stream>>>(pose, betas, rot, Ah);
  k_build_dh<<<(NTILES * KSTEPS * 32 + 255) / 256, 256, 0, stream>>>(posed, shaped, Dh);
  k_build_wh<<<(MW * 64 + 255) / 256, 256, 0, stream>>>(wgt, Wh);
  k_chain<<<2, 256, 0, stream>>>(JT, JS, betas, trans, rot, thj, Aw, A2w, out);
  k_gemm_pose<<<NTILES, 256, 0, stream>>>(Ah, Dh, vtempl, out);
  k_build_a2h<<<(BATCH * 64 + 255) / 256, 256, 0, stream>>>(A2w, A2h);
  k_skin<<<(BATCH * WTILES) / 8, 256, 0, stream>>>(Wh, A2h, out + OUT_VPOSED, trans, out);
}
